// Network_43877385896374
// MI455X (gfx1250) — matrix operand order repaired and checked by reading, not compiled, not run on hardware
//
#include <hip/hip_runtime.h>
#include <hip/hip_bf16.h>

typedef __attribute__((ext_vector_type(16))) _Float16 v16h;
typedef __attribute__((ext_vector_type(8)))  _Float16 v8h;
typedef __attribute__((ext_vector_type(8)))  float    v8f;

// Problem constants
constexpr int kL    = 4;     // layers
constexpr int kC    = 128;   // width
constexpr int kS    = 81;    // sequence (9x9)
constexpr int kSP   = 96;    // padded sequence (6 tiles of 16)
constexpr int kTB   = 128;   // T*B batch items
constexpr int kHS   = 132;   // f32 h row stride (pad to dodge bank conflicts)
constexpr int kNS   = 136;   // f16 n/att row stride (16B-aligned rows)
constexpr int kQS   = 392;   // f16 qkv row stride (384+8)
constexpr int kQKVP = kL * 24 * 4 * 32 * 16;  // packed qkv_w halfs = 196608
constexpr int kFULP = kL * 8  * 4 * 32 * 16;  // packed full_w halfs = 65536

// ---------------------------------------------------------------------------
// Prologue: convert f32 weights to f16, pre-swizzled into WMMA B-fragment
// layout: [(l, ntile, kstep)][lane][16 halfs contiguous].
// B (32x16 f16): lane = col | (khalf<<4); lanes 0-15 hold K=0..7,16..23, lanes 16-31
// hold K=8..15,24..31; half j of the v16h maps to K = khalf*8 + (j&7) + 16*(j>>3).
// ---------------------------------------------------------------------------
__global__ void pack_weights_kernel(const float* __restrict__ qkv_w,
                                    const float* __restrict__ full_w,
                                    _Float16* __restrict__ qkv_pack,
                                    _Float16* __restrict__ full_pack) {
  int idx = blockIdx.x * blockDim.x + threadIdx.x;
  if (idx < kQKVP) {
    int j    = idx & 15;
    int lane = (idx >> 4) & 31;
    int ks   = (idx >> 9) & 3;
    int t3   = idx >> 11;
    int nt   = t3 % 24;
    int l    = t3 / 24;
    int n = nt * 16 + (lane & 15);
    int k = ks * 32 + ((lane >> 4) << 3) + (j & 7) + ((j >> 3) << 4);
    qkv_pack[idx] = (_Float16)qkv_w[(l * kC + k) * (3 * kC) + n];
  } else if (idx < kQKVP + kFULP) {
    int i2   = idx - kQKVP;
    int j    = i2 & 15;
    int lane = (i2 >> 4) & 31;
    int ks   = (i2 >> 9) & 3;
    int t3   = i2 >> 11;
    int nt   = t3 % 8;
    int l    = t3 / 8;
    int n = nt * 16 + (lane & 15);
    int k = ks * 32 + ((lane >> 4) << 3) + (j & 7) + ((j >> 3) << 4);
    full_pack[idx] = (_Float16)full_w[(l * kC + k) * kC + n];
  }
}

// A fragment (16x32 f16) from LDS row-major tile:
// lane holds row m = (lane&15); halfs j=0..7 -> K = kbase+j, j=8..15 -> K = 16+kbase+(j-8),
// kbase = (lane>>4)*8.  Two contiguous 16B chunks per lane.
static __device__ inline v16h load_a_frag(const _Float16* __restrict__ base,
                                          int row, int stride, int k0) {
  const _Float16* p = base + row * stride + k0;
  v8h lo = *(const v8h*)(p);
  v8h hi = *(const v8h*)(p + 16);
  return __builtin_shufflevector(lo, hi, 0, 1, 2, 3, 4, 5, 6, 7,
                                 8, 9, 10, 11, 12, 13, 14, 15);
}

// ---------------------------------------------------------------------------
// Fully fused network: one workgroup (256 threads = 8 waves) per batch item.
// All activations live in LDS (~141 KB of the 320 KB WGP LDS).
// ---------------------------------------------------------------------------
__global__ __launch_bounds__(256) void net_fused_kernel(
    const float* __restrict__ x,     const float* __restrict__ in_w,
    const float* __restrict__ in_b,  const float* __restrict__ ln_g,
    const float* __restrict__ ln_b,  const float* __restrict__ qkv_b,
    const float* __restrict__ full_b, const float* __restrict__ out_w,
    const float* __restrict__ out_b,
    const _Float16* __restrict__ qkv_pack,
    const _Float16* __restrict__ full_pack,
    float* __restrict__ out) {
  __shared__ float    hbuf[kS * kHS];     // residual stream, f32
  __shared__ _Float16 nbuf[kSP * kNS];    // LN output, reused as attention output
  __shared__ _Float16 qkvbuf[kSP * kQS];  // q|k|v, f16

  const int tid  = threadIdx.x;
  const int lane = tid & 31;
  const int wave = tid >> 5;
  const int b    = blockIdx.x;

  // Zero f16 staging once: pad rows 81..95 stay zero forever (all later
  // writes touch rows < 81 only), so WMMA M-tiles need no predication.
  for (int i = tid; i < kSP * kNS; i += 256) nbuf[i] = (_Float16)0.f;

  // ---- input projection: h = x @ in_w + in_b  (CIN=2) ----
  for (int p = tid; p < kS * kC; p += 256) {
    int s = p >> 7, c = p & 127;
    float x0 = x[(b * kS + s) * 2 + 0];
    float x1 = x[(b * kS + s) * 2 + 1];
    hbuf[s * kHS + c] = x0 * in_w[c] + x1 * in_w[kC + c] + in_b[c];
  }
  __syncthreads();

  const int akb  = (lane >> 4) << 3;  // A-fragment K base per half-wave
  const int arow = lane & 15;         // A-fragment row within tile
  const int dcol = lane & 15;         // D-fragment col within tile
  const int drow = (lane >> 4) << 3;  // D-fragment first row within tile

  for (int l = 0; l < kL; ++l) {
    // ---- LayerNorm over C, wave-per-row, shfl_xor wave32 reduction ----
    for (int s = wave; s < kS; s += 8) {
      float v0 = hbuf[s * kHS + lane];
      float v1 = hbuf[s * kHS + lane + 32];
      float v2 = hbuf[s * kHS + lane + 64];
      float v3 = hbuf[s * kHS + lane + 96];
      float sum = v0 + v1 + v2 + v3;
      float sq  = v0 * v0 + v1 * v1 + v2 * v2 + v3 * v3;
      for (int off = 16; off > 0; off >>= 1) {
        sum += __shfl_xor(sum, off, 32);
        sq  += __shfl_xor(sq,  off, 32);
      }
      float mu   = sum * (1.0f / kC);
      float var  = sq * (1.0f / kC) - mu * mu;
      float rstd = rsqrtf(var + 1e-5f);
      const float* g = ln_g + l * kC;
      const float* bb = ln_b + l * kC;
      nbuf[s * kNS + lane]      = (_Float16)((v0 - mu) * rstd * g[lane]      + bb[lane]);
      nbuf[s * kNS + lane + 32] = (_Float16)((v1 - mu) * rstd * g[lane + 32] + bb[lane + 32]);
      nbuf[s * kNS + lane + 64] = (_Float16)((v2 - mu) * rstd * g[lane + 64] + bb[lane + 64]);
      nbuf[s * kNS + lane + 96] = (_Float16)((v3 - mu) * rstd * g[lane + 96] + bb[lane + 96]);
    }
    __syncthreads();

    // ---- qkv GEMM: [96x128] @ [128x384] via v_wmma_f32_16x16x32_f16 ----
    // 6 M-tiles x 24 N-tiles = 144 tiles, 18 per wave (wave-uniform; EXEC all-1s).
    for (int t = wave; t < 6 * 24; t += 8) {
      int mt = t % 6, nt = t / 6;
      const _Float16* bp = qkv_pack + ((l * 24 + nt) * 4) * 512 + lane * 16;
      __builtin_prefetch(bp, 0, 3);
      v8f acc = {};
      for (int ks = 0; ks < 4; ++ks) {
        v16h A = load_a_frag(nbuf, mt * 16 + arow, kNS, ks * 32 + akb);
        v16h B = *(const v16h*)(bp + ks * 512);
        acc = __builtin_amdgcn_wmma_f32_16x16x32_f16(false, A, false, B,
                                                     (short)0, acc, false, false);
      }
      int ncol   = nt * 16 + dcol;
      float bias = qkv_b[l * 384 + ncol];
      int m0     = mt * 16 + drow;
      for (int r = 0; r < 8; ++r)
        qkvbuf[(m0 + r) * kQS + ncol] = (_Float16)(acc[r] + bias);
    }
    __syncthreads();

    // ---- per-channel attention: softmax over k, head-dim 1 per channel ----
    // Two-pass (max, then fused exp-sum + weighted sum): 1 v_exp_f32 per (q,k,c).
    for (int p = tid; p < kS * kC; p += 256) {
      int q = p >> 7, c = p & 127;
      float qv = (float)qkvbuf[q * kQS + c];
      float mx = -3.402823466e38f;
      for (int k = 0; k < kS; ++k) {
        float kk = (float)qkvbuf[k * kQS + kC + c];
        mx = fmaxf(mx, qv * kk);
      }
      float se = 0.f, ac = 0.f;
      for (int k = 0; k < kS; ++k) {
        float kk = (float)qkvbuf[k * kQS + kC + c];
        float vv = (float)qkvbuf[k * kQS + 2 * kC + c];
        float e  = __expf(qv * kk - mx);
        se += e;
        ac += e * vv;
      }
      nbuf[q * kNS + c] = (_Float16)(ac / se);  // att reuses n buffer
    }
    __syncthreads();

    // ---- full GEMM [96x128]@[128x128] + bias + relu + residual into h ----
    for (int t = wave; t < 6 * 8; t += 8) {
      int mt = t % 6, nt = t / 6;
      const _Float16* bp = full_pack + ((l * 8 + nt) * 4) * 512 + lane * 16;
      __builtin_prefetch(bp, 0, 3);
      v8f acc = {};
      for (int ks = 0; ks < 4; ++ks) {
        v16h A = load_a_frag(nbuf, mt * 16 + arow, kNS, ks * 32 + akb);
        v16h B = *(const v16h*)(bp + ks * 512);
        acc = __builtin_amdgcn_wmma_f32_16x16x32_f16(false, A, false, B,
                                                     (short)0, acc, false, false);
      }
      int ncol   = nt * 16 + dcol;
      float bias = full_b[l * kC + ncol];
      int m0     = mt * 16 + drow;
      for (int r = 0; r < 8; ++r) {
        int m = m0 + r;
        if (m < kS) {
          float y = acc[r] + bias;
          hbuf[m * kHS + ncol] += (y > 0.f ? y : 0.f);
        }
      }
    }
    __syncthreads();
  }

  // ---- output projection: [81x128] @ [128x4] + out_b ----
  for (int p = tid; p < kS * 4; p += 256) {
    int s = p >> 2, j = p & 3;
    float sum = out_b[j];
    for (int c = 0; c < kC; ++c) sum += hbuf[s * kHS + c] * out_w[c * 4 + j];
    out[b * kS * 4 + p] = sum;
  }
}

extern "C" void kernel_launch(void* const* d_in, const int* in_sizes, int n_in,
                              void* d_out, int out_size, void* d_ws, size_t ws_size,
                              hipStream_t stream) {
  const float* x      = (const float*)d_in[0];
  const float* in_w   = (const float*)d_in[1];
  const float* in_b   = (const float*)d_in[2];
  const float* ln_g   = (const float*)d_in[3];
  const float* ln_b   = (const float*)d_in[4];
  const float* qkv_w  = (const float*)d_in[5];
  const float* qkv_b  = (const float*)d_in[6];
  const float* full_w = (const float*)d_in[7];
  const float* full_b = (const float*)d_in[8];
  const float* out_w  = (const float*)d_in[9];
  const float* out_b  = (const float*)d_in[10];

  _Float16* qkv_pack  = (_Float16*)d_ws;
  _Float16* full_pack = qkv_pack + kQKVP;

  // Prologue: f32 -> f16 weight pack into WMMA B-fragment layout (L2-resident).
  int pack_n = kQKVP + kFULP;
  pack_weights_kernel<<<(pack_n + 255) / 256, 256, 0, stream>>>(
      qkv_w, full_w, qkv_pack, full_pack);

  // One workgroup per batch item; all 4 layers fused, activations in LDS.
  net_fused_kernel<<<kTB, 256, 0, stream>>>(
      x, in_w, in_b, ln_g, ln_b, qkv_b, full_b, out_w, out_b,
      qkv_pack, full_pack, (float*)d_out);
}
